// DebiasedAttention_21483426414748
// MI455X (gfx1250) — matrix operand order repaired and checked by reading, not compiled, not run on hardware
//
#include <hip/hip_runtime.h>

typedef __attribute__((ext_vector_type(16))) _Float16 v16h;
typedef __attribute__((ext_vector_type(8)))  float    v8f;
typedef __attribute__((ext_vector_type(4)))  float    f32x4;
typedef __attribute__((ext_vector_type(4)))  int      i32x4;

#define S_LEN 2048
#define D_DIM 32
#define H_NUM 8
#define B_NUM 8
#define NW 4                       // waves per block (wave32)
#define SC_STRIDE (S_LEN + 4)      // pad: rows land on different banks, keeps 16B align

__global__ __launch_bounds__(128) void attn_tile_kernel(
    const float* __restrict__ q, const float* __restrict__ k,
    const float* __restrict__ v, const float* __restrict__ t,
    const int* __restrict__ mask,
    float* __restrict__ out, float* __restrict__ pattn)
{
    __shared__ float sc[16][SC_STRIDE];        // raw scores -> masked -> exp()
    __shared__ float red[16][8];
    __shared__ float rowmax[16];
    __shared__ float rowinv[16];
    __shared__ float sti[16];
    __shared__ float ored[NW][16][D_DIM];

    const int tid  = threadIdx.x;
    const int lane = tid & 31;
    const int wave = tid >> 5;
    const int n16  = lane & 15;
    const int kh   = lane >> 4;

    const int it = blockIdx.x & 127;           // S/16 tiles
    const int h  = (blockIdx.x >> 7) & 7;
    const int b  = blockIdx.x >> 10;
    const int i0 = it << 4;

    const size_t bh = (size_t)(b * H_NUM + h);
    const float* qb = q + (bh * S_LEN + (size_t)i0) * D_DIM;
    const float* kb = k + bh * S_LEN * D_DIM;
    const float* vb = v + bh * S_LEN * D_DIM;
    const float* tb = t + (size_t)b * S_LEN;
    const int*   mb = mask + ((size_t)b * S_LEN + (size_t)i0) * S_LEN;

    if (tid < 16) sti[tid] = tb[i0 + tid];

    // ---- A fragment: Q tile rows m = n16 (K index = e + 16*kh, contiguous) ----
    v16h afrag;
    {
        const float* qrow = qb + n16 * D_DIM;
        #pragma unroll
        for (int e = 0; e < 16; ++e) {
            const int kd = e + (kh << 4);
            afrag[e] = (_Float16)qrow[kd];
        }
    }
    __syncthreads();

    const float scale = 0.17677669529663687f;  // 1/sqrt(32)

    // ---- score tile sweep: S = (Q K^T)*scale * t_i*t_j  (mask applied later) ----
    for (int nt = wave; nt < (S_LEN / 16); nt += NW) {
        const int j0 = nt << 4;
        const int j  = j0 + n16;
        // B fragment: column n16 = K-row j; K index = e + 16*kh -> 64 contiguous bytes
        const f32x4* krow4 = (const f32x4*)(kb + (size_t)j * D_DIM + (kh << 4));
        const f32x4 k0 = krow4[0], k1 = krow4[1], k2 = krow4[2], k3 = krow4[3];
        v16h bfrag;
        bfrag[0]  = (_Float16)k0.x; bfrag[1]  = (_Float16)k0.y;
        bfrag[2]  = (_Float16)k0.z; bfrag[3]  = (_Float16)k0.w;
        bfrag[4]  = (_Float16)k1.x; bfrag[5]  = (_Float16)k1.y;
        bfrag[6]  = (_Float16)k1.z; bfrag[7]  = (_Float16)k1.w;
        bfrag[8]  = (_Float16)k2.x; bfrag[9]  = (_Float16)k2.y;
        bfrag[10] = (_Float16)k2.z; bfrag[11] = (_Float16)k2.w;
        bfrag[12] = (_Float16)k3.x; bfrag[13] = (_Float16)k3.y;
        bfrag[14] = (_Float16)k3.z; bfrag[15] = (_Float16)k3.w;

        v8f c = {};
        c = __builtin_amdgcn_wmma_f32_16x16x32_f16(false, afrag, false, bfrag,
                                                   (short)0, c, false, false);
        const float tjs = tb[j] * scale;
        #pragma unroll
        for (int vv = 0; vv < 8; ++vv) {
            const int m = vv + (kh << 3);
            sc[m][j] = c[vv] * tjs * sti[m];
        }
    }
    __syncthreads();

    // ---- fused mask + row max: 8 threads/row, 256 contiguous cols each ----
    const int row = tid >> 3;
    const int seg = tid & 7;
    const int c0  = seg << 8;
    {
        const i32x4* mrow = (const i32x4*)(mb + (size_t)row * S_LEN + c0);
        f32x4* srow = (f32x4*)&sc[row][c0];
        float mx = -3.4e38f;
        #pragma unroll 4
        for (int j4 = 0; j4 < 64; ++j4) {
            const i32x4 mv = mrow[j4];
            f32x4 sv = srow[j4];
            sv.x = (mv.x == 0) ? -1e9f : sv.x;
            sv.y = (mv.y == 0) ? -1e9f : sv.y;
            sv.z = (mv.z == 0) ? -1e9f : sv.z;
            sv.w = (mv.w == 0) ? -1e9f : sv.w;
            srow[j4] = sv;
            mx = fmaxf(mx, fmaxf(fmaxf(sv.x, sv.y), fmaxf(sv.z, sv.w)));
        }
        red[row][seg] = mx;
    }
    __syncthreads();
    if (seg == 0) {
        float mx = red[row][0];
        #pragma unroll
        for (int x = 1; x < 8; ++x) mx = fmaxf(mx, red[row][x]);
        rowmax[row] = mx;
    }
    __syncthreads();
    {
        const float rm = rowmax[row];
        float sum = 0.f;
        #pragma unroll 4
        for (int j = 0; j < 256; ++j) {
            const float e = __expf(sc[row][c0 + j] - rm);
            sc[row][c0 + j] = e;
            sum += e;
        }
        red[row][seg] = sum;
    }
    __syncthreads();
    if (seg == 0) {
        float sum = 0.f;
        #pragma unroll
        for (int x = 0; x < 8; ++x) sum += red[row][x];
        rowinv[row] = 1.0f / sum;
    }
    __syncthreads();

    // ---- stream p_attn out (non-temporal b128: don't pollute L2) ----
    {
        const float inv = rowinv[row];
        float* prow = pattn + (bh * S_LEN + (size_t)(i0 + row)) * S_LEN + c0;
        const float* srow = &sc[row][c0];
        #pragma unroll 4
        for (int j = 0; j < 256; j += 4) {
            f32x4 pv;
            pv.x = srow[j + 0] * inv;
            pv.y = srow[j + 1] * inv;
            pv.z = srow[j + 2] * inv;
            pv.w = srow[j + 3] * inv;
            __builtin_nontemporal_store(pv, (f32x4*)(prow + j));
        }
    }

    // ---- out = P @ V : K swept in 32-chunks, 2 N-tiles (D=32) ----
    v8f acc0 = {}, acc1 = {};
    const float invm = rowinv[n16];            // A rows: m = n16
    for (int kt = wave; kt < (S_LEN / 32); kt += NW) {
        const int kbase = kt << 5;
        v16h ap;
        #pragma unroll
        for (int e = 0; e < 16; ++e) {
            const int kd = e + (kh << 4);
            ap[e] = (_Float16)(sc[n16][kbase + kd] * invm);
        }
        v16h b0, b1;
        const float* vrow = vb + (size_t)(kbase + (kh << 4)) * D_DIM;
        #pragma unroll
        for (int e = 0; e < 16; ++e) {
            b0[e] = (_Float16)vrow[e * D_DIM + n16];
            b1[e] = (_Float16)vrow[e * D_DIM + 16 + n16];
        }
        acc0 = __builtin_amdgcn_wmma_f32_16x16x32_f16(false, ap, false, b0,
                                                      (short)0, acc0, false, false);
        acc1 = __builtin_amdgcn_wmma_f32_16x16x32_f16(false, ap, false, b1,
                                                      (short)0, acc1, false, false);
    }
    #pragma unroll
    for (int vv = 0; vv < 8; ++vv) {
        const int m = vv + (kh << 3);
        ored[wave][m][n16]      = acc0[vv];
        ored[wave][m][16 + n16] = acc1[vv];
    }
    __syncthreads();

    // ---- cross-wave reduce + store out (512 values / 128 threads) ----
    {
        float* ob = out + (bh * S_LEN + (size_t)i0) * D_DIM;
        #pragma unroll
        for (int x = 0; x < 4; ++x) {
            const int idx = tid + (x << 7);
            const int mm = idx >> 5;
            const int dd = idx & 31;
            ob[mm * D_DIM + dd] =
                ored[0][mm][dd] + ored[1][mm][dd] + ored[2][mm][dd] + ored[3][mm][dd];
        }
    }
}

extern "C" void kernel_launch(void* const* d_in, const int* in_sizes, int n_in,
                              void* d_out, int out_size, void* d_ws, size_t ws_size,
                              hipStream_t stream) {
    const float* q  = (const float*)d_in[0];
    const float* k  = (const float*)d_in[1];
    const float* v  = (const float*)d_in[2];
    const float* t  = (const float*)d_in[3];
    const int*   m  = (const int*)d_in[4];
    float* out   = (float*)d_out;
    float* pattn = out + (size_t)B_NUM * H_NUM * S_LEN * D_DIM;   // outputs concatenated

    const int blocks = B_NUM * H_NUM * (S_LEN / 16);              // 8192
    attn_tile_kernel<<<blocks, 128, 0, stream>>>(q, k, v, t, m, out, pattn);
}
